// X_attn_26834955665663
// MI455X (gfx1250) — matrix operand order repaired and checked by reading, not compiled, not run on hardware
//
#include <hip/hip_runtime.h>

// Problem constants (match reference)
#define Hdim 1024
#define Adim 512
#define Bdim 32
#define Sdim 1024
#define Tdim 1024

typedef __attribute__((ext_vector_type(16))) __bf16 v16bf;
typedef __attribute__((ext_vector_type(8)))  __bf16 v8bf;
typedef __attribute__((ext_vector_type(8)))  float  v8f;

__device__ __forceinline__ __bf16 f2bf(float f) {
  union { float f; unsigned u; } x; x.f = f;
  unsigned r = x.u + 0x7FFFu + ((x.u >> 16) & 1u);  // round-to-nearest-even
  unsigned short h = (unsigned short)(r >> 16);
  return __builtin_bit_cast(__bf16, h);
}

// Branch-free saturating tanh: 1 - 2/(exp(2x)+1).
// v_exp_f32 + v_rcp_f32, ~6 VALU ops, no EXEC divergence (vs ~60 for libm).
__device__ __forceinline__ float tanh_fast(float x) {
  float t = __expf(2.0f * x);            // exp(2x); 0 at -inf, inf at +inf
  return 1.0f - 2.0f * __builtin_amdgcn_rcpf(t + 1.0f);
}

// ---------------------------------------------------------------- utilities
__global__ void cvt_bf16_kernel(const float* __restrict__ src,
                                __bf16* __restrict__ dst, int n) {
  int i = blockIdx.x * blockDim.x + threadIdx.x;
  int stride = gridDim.x * blockDim.x;
  for (; i < n; i += stride) dst[i] = f2bf(src[i]);
}

__global__ void zero_kernel(float* __restrict__ p, int n) {
  int i = blockIdx.x * blockDim.x + threadIdx.x;
  int stride = gridDim.x * blockDim.x;
  for (; i < n; i += stride) p[i] = 0.0f;
}

// --------------------------------------------------- fused WMMA score kernel
// e[row] = Sum_a w4[a] * tanh( X[row,:] . Wb[a,:] )  (+ optional biases)
// X: [M,H] fp32 (rows flat (s*B+b)), Wb: [A,H] bf16, w4: [A]
// Grid: M/(MT*16) blocks of 128 threads (4 waves). Each block stages MT*16
// rows in LDS (bf16); each wave owns 8 of the 32 N-tiles; every B fragment
// loaded from L2 is reused by MT WMMAs (cuts weight-stream traffic by MT).
template <int MT>
__global__ __launch_bounds__(128)
void score_wmma_kernel(const float* __restrict__ X,
                       const __bf16* __restrict__ Wb,
                       const float* __restrict__ w4,
                       const float* __restrict__ biasB,
                       const float* __restrict__ biasB2,
                       const float* __restrict__ rowExtra,
                       float* __restrict__ outE) {
  extern __shared__ char smem[];
  float*  e_sm = (float*)smem;            // MT*16 floats
  __bf16* Xs   = (__bf16*)(smem + 256);   // MT*16 x Hdim bf16

  const int tid = threadIdx.x;
  const int r0  = blockIdx.x * (MT * 16);

  // Stage MT*16 rows of X into LDS as bf16 (coalesced fp32 reads + convert)
  for (int i = tid; i < MT * 16 * Hdim; i += 128)
    Xs[i] = f2bf(X[(size_t)r0 * Hdim + i]);
  if (tid < MT * 16) e_sm[tid] = 0.0f;
  __syncthreads();

  const int lane = tid & 31;
  const int wave = tid >> 5;
  const int nlo  = lane & 15;   // N index within tile / M index for A frag
  const int half = lane >> 4;   // lane-group select

  float eacc[MT][8];
#pragma unroll
  for (int m = 0; m < MT; ++m)
#pragma unroll
    for (int r = 0; r < 8; ++r) eacc[m][r] = 0.0f;

  for (int nt = wave; nt < Adim / 16; nt += 4) {
    v8f acc[MT];
#pragma unroll
    for (int m = 0; m < MT; ++m)
#pragma unroll
      for (int j = 0; j < 8; ++j) acc[m][j] = 0.0f;

    const __bf16* wbase = Wb + (size_t)(nt * 16 + nlo) * Hdim;

#pragma unroll 2
    for (int k = 0; k < Hdim / 32; ++k) {
      // B fragment 32x16 bf16: lane n=nlo, element j -> K=32k+16*half+j
      v16bf bfrag = *(const v16bf*)(wbase + k * 32 + half * 16);
#pragma unroll
      for (int m = 0; m < MT; ++m) {
        // A fragment 16x32 bf16: lane row=nlo, elem j -> K=32k+16*half+j
        const __bf16* xbase = &Xs[(m * 16 + nlo) * Hdim + k * 32 + half * 16];
        v8bf alo = *(const v8bf*)(xbase);
        v8bf ahi = *(const v8bf*)(xbase + 8);
        v16bf afrag;
#pragma unroll
        for (int j = 0; j < 8; ++j) { afrag[j] = alo[j]; afrag[j + 8] = ahi[j]; }
        acc[m] = __builtin_amdgcn_wmma_f32_16x16x32_bf16(
            false, afrag, false, bfrag, (short)0, acc[m], false, false);
      }
    }

    // Fused epilogue: sum_a w4[a]*tanh(acc); this lane holds column a=16*nt+nlo
    float wn = w4[nt * 16 + nlo];
#pragma unroll
    for (int m = 0; m < MT; ++m)
#pragma unroll
      for (int r = 0; r < 8; ++r) eacc[m][r] += wn * tanh_fast(acc[m][r]);
  }

  // Reduce across the 16 N-lanes of each half (C layout: row m = r + 8*half)
#pragma unroll
  for (int m = 0; m < MT; ++m) {
#pragma unroll
    for (int r = 0; r < 8; ++r) {
      float v = eacc[m][r];
      v += __shfl_xor(v, 1);
      v += __shfl_xor(v, 2);
      v += __shfl_xor(v, 4);
      v += __shfl_xor(v, 8);
      eacc[m][r] = v;
    }
    if (nlo == 0) {
#pragma unroll
      for (int r = 0; r < 8; ++r)
        atomicAdd(&e_sm[m * 16 + r + 8 * half], eacc[m][r]);
    }
  }
  __syncthreads();

  if (tid < MT * 16) {
    int row = r0 + tid;
    int b = row & (Bdim - 1);
    float e = e_sm[tid];
    if (biasB)    e += biasB[b];
    if (biasB2)   e += biasB2[b];
    if (rowExtra) e += rowExtra[row];
    outE[row] = e;
  }
}

// -------- per-row coverage term: Sum_a w4c[a]*tanh(ax[b,s]*W3[a]), row=s*B+b
__global__ __launch_bounds__(256)
void axterm_kernel(const float* __restrict__ ax, const float* __restrict__ W3,
                   const float* __restrict__ w4c, float* __restrict__ rowExtra) {
  int lane = threadIdx.x & 31;
  int wv   = threadIdx.x >> 5;
  int row  = blockIdx.x * 8 + wv;      // row = s*B + b
  int b = row & (Bdim - 1);
  int s = row >> 5;
  float av = ax[b * Sdim + s];
  float acc = 0.0f;
  for (int a = lane; a < Adim; a += 32)
    acc += w4c[a] * tanh_fast(av * W3[a]);
  acc += __shfl_xor(acc, 1);
  acc += __shfl_xor(acc, 2);
  acc += __shfl_xor(acc, 4);
  acc += __shfl_xor(acc, 8);
  acc += __shfl_xor(acc, 16);
  if (lane == 0) rowExtra[row] = acc;
}

// -------- per-b softmax over len values; e indexed [s*B+b], p out [b*len+s]
__global__ __launch_bounds__(256)
void softmax_kernel(const float* __restrict__ e, float* __restrict__ p,
                    const float* __restrict__ axOld, float* __restrict__ axNewOut,
                    int len) {
  __shared__ float red[256];
  const int b = blockIdx.x, tid = threadIdx.x;
  float vals[4];
  float mx = -3.4e38f;
#pragma unroll
  for (int i = 0; i < 4; ++i) {
    int s = tid + i * 256;
    vals[i] = e[s * Bdim + b];
    mx = fmaxf(mx, vals[i]);
  }
  red[tid] = mx; __syncthreads();
  for (int off = 128; off > 0; off >>= 1) {
    if (tid < off) red[tid] = fmaxf(red[tid], red[tid + off]);
    __syncthreads();
  }
  mx = red[0]; __syncthreads();
  float sum = 0.0f;
#pragma unroll
  for (int i = 0; i < 4; ++i) { vals[i] = __expf(vals[i] - mx); sum += vals[i]; }
  red[tid] = sum; __syncthreads();
  for (int off = 128; off > 0; off >>= 1) {
    if (tid < off) red[tid] += red[tid + off];
    __syncthreads();
  }
  float inv = 1.0f / red[0];
#pragma unroll
  for (int i = 0; i < 4; ++i) {
    int s = tid + i * 256;
    float pv = vals[i] * inv;
    p[b * len + s] = pv;
    if (axNewOut) axNewOut[b * len + s] = axOld[b * len + s] + pv;
  }
}

// -------- context: ctx[b,h] += Sum_{s in slice} p[b,s] * Xbig[s,b,h]
__global__ __launch_bounds__(256)
void ctx_kernel(const float* __restrict__ p, const float* __restrict__ Xbig,
                float* __restrict__ ctx) {
  int h  = blockIdx.x * 256 + threadIdx.x;
  int b  = blockIdx.y;
  int s0 = blockIdx.z * 128;
  float acc = 0.0f;
  for (int s = s0; s < s0 + 128; ++s)
    acc += p[b * Sdim + s] * Xbig[((size_t)s * Bdim + b) * Hdim + h];
  atomicAdd(&ctx[b * Hdim + h], acc);
}

// -------- zi[b] = sigmoid([cxi,cti] . W11)
__global__ __launch_bounds__(256)
void zi_kernel(const float* __restrict__ cxi, const float* __restrict__ cti,
               const float* __restrict__ W11, float* __restrict__ zi) {
  __shared__ float red[256];
  const int b = blockIdx.x, tid = threadIdx.x;
  float acc = 0.0f;
  for (int j = tid; j < Hdim; j += 256)
    acc += cxi[b * Hdim + j] * W11[j] + cti[b * Hdim + j] * W11[Hdim + j];
  red[tid] = acc; __syncthreads();
  for (int off = 128; off > 0; off >>= 1) {
    if (tid < off) red[tid] += red[tid + off];
    __syncthreads();
  }
  if (tid == 0) zi[b] = 1.0f / (1.0f + __expf(-red[0]));
}

// -------- ci = z*tanh(cxi@W9^T) + (1-z)*tanh(cti@W10^T)
__global__ __launch_bounds__(256)
void gate_kernel(const float* __restrict__ cxi, const float* __restrict__ cti,
                 const float* __restrict__ W9, const float* __restrict__ W10,
                 const float* __restrict__ zi, float* __restrict__ out) {
  int h = blockIdx.x * 256 + threadIdx.x;
  int b = blockIdx.y;
  const float* cx  = cxi + b * Hdim;
  const float* ct  = cti + b * Hdim;
  const float* w9  = W9  + (size_t)h * Hdim;
  const float* w10 = W10 + (size_t)h * Hdim;
  float gx = 0.0f, gt = 0.0f;
  for (int k = 0; k < Hdim; ++k) { gx += cx[k] * w9[k]; gt += ct[k] * w10[k]; }
  float z = zi[b];
  out[b * Hdim + h] = z * tanh_fast(gx) + (1.0f - z) * tanh_fast(gt);
}

// ---------------------------------------------------------------- dispatcher
extern "C" void kernel_launch(void* const* d_in, const int* in_sizes, int n_in,
                              void* d_out, int out_size, void* d_ws, size_t ws_size,
                              hipStream_t stream) {
  const float* hidden = (const float*)d_in[0];
  const float* enc    = (const float*)d_in[1];
  const float* h0s    = (const float*)d_in[2];
  const float* ax     = (const float*)d_in[3];
  const float* W1  = (const float*)d_in[4];
  const float* W2  = (const float*)d_in[5];
  const float* W3  = (const float*)d_in[6];
  const float* W4  = (const float*)d_in[7];
  const float* W5  = (const float*)d_in[8];
  const float* W6  = (const float*)d_in[9];
  const float* W7  = (const float*)d_in[10];
  const float* W8  = (const float*)d_in[11];
  const float* W9  = (const float*)d_in[12];
  const float* W10 = (const float*)d_in[13];
  const float* W11 = (const float*)d_in[14];
  float* out = (float*)d_out;  // [B*H] ci, then [B*S] ax_new

  // Workspace carve-out (256B aligned slices)
  char* ws = (char*)d_ws;
  size_t off = 0;
  auto alloc = [&](size_t bytes) -> char* {
    char* p = ws + off; off += (bytes + 255) & ~(size_t)255; return p;
  };
  __bf16* W1b = (__bf16*)alloc((size_t)Adim * Hdim * 2);
  __bf16* W2b = (__bf16*)alloc((size_t)Adim * Hdim * 2);
  __bf16* W5b = (__bf16*)alloc((size_t)Adim * Hdim * 2);
  __bf16* W6b = (__bf16*)alloc((size_t)Adim * Hdim * 2);
  __bf16* W7b = (__bf16*)alloc((size_t)Adim * Hdim * 2);
  float* sxy  = (float*)alloc(Bdim * 4);
  float* sty  = (float*)alloc(Bdim * 4);
  float* stx  = (float*)alloc(Bdim * 4);
  float* ziB  = (float*)alloc(Bdim * 4);
  float* rowX = (float*)alloc((size_t)Sdim * Bdim * 4);
  float* ex   = (float*)alloc((size_t)Sdim * Bdim * 4);
  float* et   = (float*)alloc((size_t)Tdim * Bdim * 4);
  float* px   = (float*)alloc((size_t)Bdim * Sdim * 4);
  float* pt   = (float*)alloc((size_t)Bdim * Tdim * 4);
  float* cxi  = (float*)alloc((size_t)Bdim * Hdim * 4);
  float* cti  = (float*)alloc((size_t)Bdim * Hdim * 4);
  (void)ws_size; (void)in_sizes; (void)n_in; (void)out_size;

  const int nW = Adim * Hdim;
  const size_t sh1 = 256 + (size_t)1 * 16 * Hdim * 2;   //  32.25 KB
  const size_t sh4 = 256 + (size_t)4 * 16 * Hdim * 2;   // 128.25 KB

  // 1) bf16 weight staging (L2-resident, reused by every score block)
  cvt_bf16_kernel<<<512, 256, 0, stream>>>(W1, W1b, nW);
  cvt_bf16_kernel<<<512, 256, 0, stream>>>(W2, W2b, nW);
  cvt_bf16_kernel<<<512, 256, 0, stream>>>(W5, W5b, nW);
  cvt_bf16_kernel<<<512, 256, 0, stream>>>(W6, W6b, nW);
  cvt_bf16_kernel<<<512, 256, 0, stream>>>(W7, W7b, nW);

  // 2) per-b broadcast constants: sxy = red(tanh(h@W2^T))·W4[A:2A],
  //                                sty = red(tanh(h@W7^T))·W8[2A:3A]
  score_wmma_kernel<1><<<Bdim / 16, 128, sh1, stream>>>(
      hidden, W2b, W4 + Adim, nullptr, nullptr, nullptr, sxy);
  score_wmma_kernel<1><<<Bdim / 16, 128, sh1, stream>>>(
      hidden, W7b, W8 + 2 * Adim, nullptr, nullptr, nullptr, sty);

  // 3) coverage term per (s,b)
  axterm_kernel<<<(Sdim * Bdim) / 8, 256, 0, stream>>>(ax, W3, W4 + 2 * Adim, rowX);

  // 4) encoder scores (big WMMA pass, 64-row tiles) -> softmax -> cxi, ax_new
  score_wmma_kernel<4><<<(Sdim * Bdim) / 64, 128, sh4, stream>>>(
      enc, W1b, W4, sxy, nullptr, rowX, ex);
  softmax_kernel<<<Bdim, 256, 0, stream>>>(ex, px, ax, out + Bdim * Hdim, Sdim);
  zero_kernel<<<128, 256, 0, stream>>>(cxi, Bdim * Hdim);
  ctx_kernel<<<dim3(Hdim / 256, Bdim, 8), 256, 0, stream>>>(px, enc, cxi);

  // 5) stx = red(tanh(cxi@W6^T))·W8[A:2A]
  score_wmma_kernel<1><<<Bdim / 16, 128, sh1, stream>>>(
      cxi, W6b, W8 + Adim, nullptr, nullptr, nullptr, stx);

  // 6) decoder scores (big WMMA pass) -> softmax -> cti
  score_wmma_kernel<4><<<(Tdim * Bdim) / 64, 128, sh4, stream>>>(
      h0s, W5b, W8, sty, stx, nullptr, et);
  softmax_kernel<<<Bdim, 256, 0, stream>>>(et, pt, nullptr, nullptr, Tdim);
  zero_kernel<<<128, 256, 0, stream>>>(cti, Bdim * Hdim);
  ctx_kernel<<<dim3(Hdim / 256, Bdim, 8), 256, 0, stream>>>(pt, h0s, cti);

  // 7) gated fusion
  zi_kernel<<<Bdim, 256, 0, stream>>>(cxi, cti, W11, ziB);
  gate_kernel<<<dim3(Hdim / 256, Bdim), 256, 0, stream>>>(cxi, cti, W9, W10, ziB, out);
}
